// TextDecoderHeadWithCodebook_33715493273610
// MI455X (gfx1250) — matrix operand order repaired and checked by reading, not compiled, not run on hardware
//
#include <hip/hip_runtime.h>

typedef __attribute__((ext_vector_type(8)))  __bf16 v8bf;
typedef __attribute__((ext_vector_type(16))) __bf16 v16bf;
typedef __attribute__((ext_vector_type(8)))  float  v8f;

#define BK 64
#define BM 256
#define BN 128
#define STRIDE 72          // 64 + 8 bf16 pad; 144B row pitch, 16B aligned
#define USE_ASYNC_STAGE 1  // CDNA5 async global->LDS staging (ASYNCcnt)

// ---------------------------------------------------------------- fragment helpers

__device__ __forceinline__ v16bf make_frag(v8bf lo, v8bf hi) {
  return __builtin_shufflevector(lo, hi, 0,1,2,3,4,5,6,7,8,9,10,11,12,13,14,15);
}

// A-matrix 16x32 bf16 fragment (lane m, half0 -> K0..7 & K16..23)
__device__ __forceinline__ v16bf load_a_frag(const __bf16* tile, int lane) {
  int r = lane & 15, h = (lane >> 4) & 1;
  const __bf16* p = tile + r * STRIDE + h * 8;
  v8bf lo = *(const v8bf*)p;
  v8bf hi = *(const v8bf*)(p + 16);
  return make_frag(lo, hi);
}

// B-matrix 32x16 bf16 fragment (lane n, half0 -> K0..7 & K16..23)
__device__ __forceinline__ v16bf load_b_frag(const __bf16* tile, int lane) {
  int c = lane & 15, h = (lane >> 4) & 1;
  const __bf16* p = tile + c * STRIDE + h * 8;
  v8bf lo = *(const v8bf*)p;
  v8bf hi = *(const v8bf*)(p + 16);
  return make_frag(lo, hi);
}

// ---------------------------------------------------------------- tile staging (ROWS x 64 bf16)
// ROWS*8 chunks of 16B; 256 threads -> ROWS/32 chunks per thread.

template <int ROWS>
__device__ __forceinline__ void stage_tile(const __bf16* __restrict__ g, int ld,
                                           int row0, int k0, __bf16* lds, int tid) {
  #pragma unroll
  for (int i = 0; i < ROWS / 32; ++i) {
    int c = tid + 256 * i;
    int row = c >> 3, col = c & 7;  // col in 16B units
    const __bf16* gp = g + (long)(row0 + row) * ld + k0 + col * 8;
    __bf16* lp = lds + row * STRIDE + col * 8;
#if USE_ASYNC_STAGE
    unsigned loff = (unsigned)(size_t)(void*)lp;  // LDS aperture: low 32 bits = DS byte addr
    asm volatile("global_load_async_to_lds_b128 %0, %1, off"
                 :: "v"(loff), "v"((unsigned long long)(size_t)gp)
                 : "memory");
#else
    *(v8bf*)lp = *(const v8bf*)gp;
#endif
  }
}

__device__ __forceinline__ void wait_stage() {
#if USE_ASYNC_STAGE
  asm volatile("s_wait_asynccnt 0x0" ::: "memory");
#endif
  __syncthreads();
}

// ---------------------------------------------------------------- misc helpers

__device__ __forceinline__ unsigned long long shfl_xor_u64(unsigned long long v, int mask) {
  int lo = __shfl_xor((int)(unsigned)(v & 0xffffffffull), mask, 32);
  int hi = __shfl_xor((int)(unsigned)(v >> 32), mask, 32);
  return ((unsigned long long)(unsigned)hi << 32) | (unsigned)lo;
}

// total-order float key (smaller float -> smaller key), packed with index
__device__ __forceinline__ unsigned long long pack_key(float d, int n) {
  unsigned u = __float_as_uint(d);
  u = (u & 0x80000000u) ? ~u : (u | 0x80000000u);
  return ((unsigned long long)u << 32) | (unsigned)n;
}

__device__ __forceinline__ float block_sum(float v, float* red8) {
  #pragma unroll
  for (int m = 16; m >= 1; m >>= 1) v += __shfl_xor(v, m, 32);
  int wid = threadIdx.x >> 5;
  if ((threadIdx.x & 31) == 0) red8[wid] = v;
  __syncthreads();
  float t = 0.f;
  #pragma unroll
  for (int w = 0; w < 8; ++w) t += red8[w];
  __syncthreads();
  return t;
}

// ---------------------------------------------------------------- shared WMMA main loop
// Block tile BM(256) x BN(128); 8 waves in 4(M) x 2(N) grid; wave tile 64x64.
// acc[4][4] = 128 VGPRs of f32 accumulators per lane.

__device__ __forceinline__ void gemm_mainloop(const __bf16* __restrict__ A,
                                              const __bf16* __restrict__ B,
                                              int K, int lda, int ldb,
                                              int rowA0, int rowB0,
                                              __bf16 (*sA)[BM * STRIDE],
                                              __bf16 (*sB)[BN * STRIDE],
                                              v8f acc[4][4]) {
  int tid = threadIdx.x;
  int lane = tid & 31, wid = tid >> 5;
  int waveM = wid & 3, waveN = wid >> 2;

  int nst = K / BK;
  stage_tile<BM>(A, lda, rowA0, 0, sA[0], tid);
  stage_tile<BN>(B, ldb, rowB0, 0, sB[0], tid);

  int cur = 0;
  for (int s = 0; s < nst; ++s) {
    wait_stage();  // own ASYNCcnt drained + barrier => buf[cur] valid block-wide
    if (s + 1 < nst) {
      stage_tile<BM>(A, lda, rowA0, (s + 1) * BK, sA[cur ^ 1], tid);
      stage_tile<BN>(B, ldb, rowB0, (s + 1) * BK, sB[cur ^ 1], tid);
    }
    const __bf16* bA = sA[cur];
    const __bf16* bB = sB[cur];
    #pragma unroll
    for (int kk = 0; kk < 2; ++kk) {
      v16bf bf[4];
      #pragma unroll
      for (int j = 0; j < 4; ++j)
        bf[j] = load_b_frag(bB + kk * 32 + (waveN * 64 + j * 16) * STRIDE, lane);
      #pragma unroll
      for (int i = 0; i < 4; ++i) {
        v16bf af = load_a_frag(bA + kk * 32 + (waveM * 64 + i * 16) * STRIDE, lane);
        #pragma unroll
        for (int j = 0; j < 4; ++j)
          acc[i][j] = __builtin_amdgcn_wmma_f32_16x16x32_bf16(
              false, af, false, bf[j], (short)0, acc[i][j], false, false);
      }
    }
    cur ^= 1;
  }
}

// ---------------------------------------------------------------- GEMM + bias -> bf16
// C[M x N] = A[M x K] * B[N x K]^T + bias ; grid (N/BN, M/BM), 256 threads
__global__ __launch_bounds__(256) void gemm_bias_kernel(
    const __bf16* __restrict__ A, const __bf16* __restrict__ B,
    const float* __restrict__ bias, __bf16* __restrict__ C,
    int K, int lda, int ldb, int ldc) {
  __shared__ __bf16 sA[2][BM * STRIDE];
  __shared__ __bf16 sB[2][BN * STRIDE];
  v8f acc[4][4] = {};

  gemm_mainloop(A, B, K, lda, ldb, blockIdx.y * BM, blockIdx.x * BN, sA, sB, acc);

  int lane = threadIdx.x & 31, wid = threadIdx.x >> 5;
  int waveM = wid & 3, waveN = wid >> 2;
  int colw = lane & 15, hsel = lane >> 4;
  #pragma unroll
  for (int j = 0; j < 4; ++j) {
    int col = blockIdx.x * BN + waveN * 64 + j * 16 + colw;
    float bv = bias[col];
    #pragma unroll
    for (int i = 0; i < 4; ++i) {
      long rbase = blockIdx.y * BM + waveM * 64 + i * 16 + hsel * 8;
      #pragma unroll
      for (int v = 0; v < 8; ++v)
        C[(rbase + v) * (long)ldc + col] = (__bf16)(acc[i][j][v] + bv);
    }
  }
}

// ---------------------------------------------------------------- distance GEMM + argmin
__global__ __launch_bounds__(256) void argmin_kernel(
    const __bf16* __restrict__ Z, const __bf16* __restrict__ CB,
    const float* __restrict__ znorm, const float* __restrict__ cbnorm,
    unsigned long long* __restrict__ best, int Kd, int lda, int ldb) {
  __shared__ __bf16 sA[2][BM * STRIDE];
  __shared__ __bf16 sB[2][BN * STRIDE];
  v8f acc[4][4] = {};

  gemm_mainloop(Z, CB, Kd, lda, ldb, blockIdx.y * BM, blockIdx.x * BN, sA, sB, acc);

  int lane = threadIdx.x & 31, wid = threadIdx.x >> 5;
  int waveM = wid & 3, waveN = wid >> 2;
  int colw = lane & 15, hsel = lane >> 4;

  int n[4];
  float cn[4];
  #pragma unroll
  for (int j = 0; j < 4; ++j) {
    n[j] = blockIdx.x * BN + waveN * 64 + j * 16 + colw;
    cn[j] = cbnorm[n[j]];
  }

  #pragma unroll
  for (int i = 0; i < 4; ++i) {
    int rbase = blockIdx.y * BM + waveM * 64 + i * 16 + hsel * 8;
    #pragma unroll
    for (int v = 0; v < 8; ++v) {
      int m = rbase + v;
      float zn = znorm[m];
      unsigned long long pm = ~0ull;
      #pragma unroll
      for (int j = 0; j < 4; ++j) {
        unsigned long long p = pack_key(zn + cn[j] - 2.0f * acc[i][j][v], n[j]);
        if (p < pm) pm = p;
      }
      #pragma unroll
      for (int msk = 1; msk <= 8; msk <<= 1) {
        unsigned long long o = shfl_xor_u64(pm, msk);
        if (o < pm) pm = o;
      }
      if (colw == 0) atomicMin(best + m, pm);
    }
  }
}

// ---------------------------------------------------------------- f32 -> bf16 bulk convert
__global__ __launch_bounds__(256) void cvt_bf16_kernel(const float* __restrict__ src,
                                                       __bf16* __restrict__ dst) {
  long i = ((long)blockIdx.x * 256 + threadIdx.x) * 8;
  float4 a = *(const float4*)(src + i);
  float4 b = *(const float4*)(src + i + 4);
  v8bf v;
  v[0]=(__bf16)a.x; v[1]=(__bf16)a.y; v[2]=(__bf16)a.z; v[3]=(__bf16)a.w;
  v[4]=(__bf16)b.x; v[5]=(__bf16)b.y; v[6]=(__bf16)b.z; v[7]=(__bf16)b.w;
  *(v8bf*)(dst + i) = v;
}

// ---------------------------------------------------------------- LayerNorm (+GELU) in place
template <int D, bool GELU>
__global__ __launch_bounds__(256) void ln_kernel(
    __bf16* __restrict__ X, const float* __restrict__ g, const float* __restrict__ be,
    float* __restrict__ znorm, unsigned long long* __restrict__ best) {
  constexpr int PT = D / 256;  // 16 or 8
  __shared__ float red8[8];

  int row = blockIdx.x, tid = threadIdx.x;
  __bf16* p = X + (long)row * D + tid * PT;

  float x[PT];
  #pragma unroll
  for (int c = 0; c < PT / 8; ++c) {
    v8bf v = *(const v8bf*)(p + c * 8);
    #pragma unroll
    for (int e = 0; e < 8; ++e) x[c * 8 + e] = (float)v[e];
  }

  float s = 0.f, q = 0.f;
  #pragma unroll
  for (int e = 0; e < PT; ++e) { s += x[e]; q += x[e] * x[e]; }
  float ts = block_sum(s, red8);
  float tq = block_sum(q, red8);
  float mu = ts / (float)D;
  float var = tq / (float)D - mu * mu;
  float rstd = rsqrtf(var + 1e-5f);

  float qn = 0.f;
  #pragma unroll
  for (int c = 0; c < PT / 8; ++c) {
    v8bf v;
    #pragma unroll
    for (int e = 0; e < 8; ++e) {
      int col = tid * PT + c * 8 + e;
      float y = (x[c * 8 + e] - mu) * rstd * g[col] + be[col];
      if (GELU) y = 0.5f * y * (1.0f + erff(y * 0.70710678118654752f));
      __bf16 yb = (__bf16)y;
      v[e] = yb;
      if (!GELU) { float yr = (float)yb; qn += yr * yr; }
    }
    *(v8bf*)(p + c * 8) = v;
  }

  if (!GELU) {
    float tn = block_sum(qn, red8);
    if (tid == 0) {
      znorm[row] = tn;
      best[row] = ~0ull;
    }
  }
}

// ---------------------------------------------------------------- codebook norms (bf16-rounded)
__global__ __launch_bounds__(256) void cbnorm_kernel(const float* __restrict__ CB,
                                                     float* __restrict__ cbn) {
  __shared__ float red8[8];
  int row = blockIdx.x, tid = threadIdx.x;
  const float4* p = (const float4*)(CB + (long)row * 2048) + tid * 2;
  float4 a = p[0], b = p[1];
  float q = 0.f, r;
  r = (float)(__bf16)a.x; q += r * r;  r = (float)(__bf16)a.y; q += r * r;
  r = (float)(__bf16)a.z; q += r * r;  r = (float)(__bf16)a.w; q += r * r;
  r = (float)(__bf16)b.x; q += r * r;  r = (float)(__bf16)b.y; q += r * r;
  r = (float)(__bf16)b.z; q += r * r;  r = (float)(__bf16)b.w; q += r * r;
  float t = block_sum(q, red8);
  if (tid == 0) cbn[row] = t;
}

// ---------------------------------------------------------------- final gather: out = codebook[idx]
__global__ __launch_bounds__(256) void gather_kernel(const float* __restrict__ CB,
                                                     const unsigned long long* __restrict__ best,
                                                     float* __restrict__ out) {
  int row = blockIdx.x, tid = threadIdx.x;
  unsigned idx = (unsigned)(best[row] & 0xffffffffull);
  const float4* s = (const float4*)(CB + (long)idx * 2048) + tid * 2;
  float4* d = (float4*)(out + (long)row * 2048) + tid * 2;
  d[0] = s[0];
  d[1] = s[1];
}

// ---------------------------------------------------------------- launch

extern "C" void kernel_launch(void* const* d_in, const int* in_sizes, int n_in,
                              void* d_out, int out_size, void* d_ws, size_t ws_size,
                              hipStream_t stream) {
  const float* latents = (const float*)d_in[0];   // [8192 x 1536]
  const float* W1      = (const float*)d_in[1];   // [4096 x 1536]
  const float* b1      = (const float*)d_in[2];
  const float* g1      = (const float*)d_in[3];
  const float* be1     = (const float*)d_in[4];
  const float* W2      = (const float*)d_in[5];   // [2048 x 4096]
  const float* b2      = (const float*)d_in[6];
  const float* g2      = (const float*)d_in[7];
  const float* be2     = (const float*)d_in[8];
  const float* cb      = (const float*)d_in[9];   // [8192 x 2048]
  float* out = (float*)d_out;

  const int M = 8192, Dh = 4096, Din = 1536, Dout = 2048, Kc = 8192;

  char* ws = (char*)d_ws;
  size_t o = 0;
  __bf16* h      = (__bf16*)(ws + o); o += (size_t)M * Dh * 2;     // 64 MB
  __bf16* z      = (__bf16*)(ws + o); o += (size_t)M * Dout * 2;   // 32 MB
  __bf16* lat_b  = (__bf16*)(ws + o); o += (size_t)M * Din * 2;    // 24 MB
  __bf16* W1_b   = (__bf16*)(ws + o); o += (size_t)Dh * Din * 2;   // 12 MB
  __bf16* W2_b   = (__bf16*)(ws + o); o += (size_t)Dout * Dh * 2;  // 16 MB
  __bf16* cb_b   = (__bf16*)(ws + o); o += (size_t)Kc * Dout * 2;  // 32 MB
  float* znorm   = (float*)(ws + o);  o += (size_t)M * 4;
  float* cbn     = (float*)(ws + o);  o += (size_t)Kc * 4;
  unsigned long long* best = (unsigned long long*)(ws + o);

  // one-shot operand conversions to bf16 (enables pure async-to-LDS staging)
  cvt_bf16_kernel<<<(int)((size_t)M * Din / 2048), 256, 0, stream>>>(latents, lat_b);
  cvt_bf16_kernel<<<(int)((size_t)Dh * Din / 2048), 256, 0, stream>>>(W1, W1_b);
  cvt_bf16_kernel<<<(int)((size_t)Dout * Dh / 2048), 256, 0, stream>>>(W2, W2_b);
  cvt_bf16_kernel<<<(int)((size_t)Kc * Dout / 2048), 256, 0, stream>>>(cb, cb_b);
  cbnorm_kernel<<<Kc, 256, 0, stream>>>(cb, cbn);

  // h = GELU(LN(latents @ W1^T + b1))
  gemm_bias_kernel<<<dim3(Dh / BN, M / BM), 256, 0, stream>>>(
      lat_b, W1_b, b1, h, Din, Din, Din, Dh);
  ln_kernel<4096, true><<<M, 256, 0, stream>>>(h, g1, be1, nullptr, nullptr);

  // z = LN(h @ W2^T + b2) ; also |z|^2 and argmin-table init
  gemm_bias_kernel<<<dim3(Dout / BN, M / BM), 256, 0, stream>>>(
      h, W2_b, b2, z, Dh, Dh, Dh, Dout);
  ln_kernel<2048, false><<<M, 256, 0, stream>>>(z, g2, be2, znorm, best);

  // nearest codeword per row
  argmin_kernel<<<dim3(Kc / BN, M / BM), 256, 0, stream>>>(
      z, cb_b, znorm, cbn, best, Dout, Dout, Dout);

  // straight-through forward value = codebook[idx]
  gather_kernel<<<M, 256, 0, stream>>>(cb, best, out);
}
